// MultiheadedSelfAttention_56453050138905
// MI455X (gfx1250) — matrix operand order repaired and checked by reading, not compiled, not run on hardware
//
#include <hip/hip_runtime.h>
#include <hip/hip_bf16.h>

typedef __attribute__((ext_vector_type(16))) __bf16 v16bf;
typedef __attribute__((ext_vector_type(8)))  __bf16 v8bf;
typedef __attribute__((ext_vector_type(8)))  float  v8f;

#define C_EMB   1024
#define HEADS   16
#define HD      64
#define BATCH   2
#define SEQ     2048
#define ROWS    (BATCH*SEQ)      /* 4096 */
#define C3      (3*C_EMB)        /* 3072 */
#define P_LDS_STRIDE 72          /* 16-byte aligned row stride (144B) for P staging */

static __device__ __forceinline__ v8f wmma_bf16(v16bf a, v16bf b, v8f c) {
  // emits v_wmma_f32_16x16x32_bf16
  return __builtin_amdgcn_wmma_f32_16x16x32_bf16(
      /*neg_a=*/false, a, /*neg_b=*/false, b,
      /*c_mod=*/(short)0, c, /*reuse_a=*/false, /*reuse_b=*/false);
}

// A-fragment (16x32 bf16, MxK) from row-major [lda] memory.
// Lane L (m = L&15, half = L>>4) holds K = {k0+half*8 .. +8} U {k0+16+half*8 .. +8}.
static __device__ __forceinline__ v16bf load_a_frag(const __bf16* base, int lda,
                                                    int row0, int k0, int lane) {
  int m = lane & 15, half = lane >> 4;
  const __bf16* p = base + (size_t)(row0 + m) * lda + k0 + half * 8;
  union { v16bf v; v8bf h[2]; } u;
  u.h[0] = *(const v8bf*)(p);
  u.h[1] = *(const v8bf*)(p + 16);
  return u.v;
}

// B-fragment (32x16 bf16, KxN) loaded from B-transposed row-major [ldbt] memory.
// Lane L (n = L&15, half = L>>4) holds K = {k0+half*8 .. +8} U {k0+16+half*8 .. +8}.
static __device__ __forceinline__ v16bf load_bt_frag(const __bf16* baseT, int ldbt,
                                                     int n0, int k0, int lane) {
  int n = lane & 15, half = lane >> 4;
  const __bf16* p = baseT + (size_t)(n0 + n) * ldbt + k0 + half * 8;
  union { v16bf v; v8bf h[2]; } u;
  u.h[0] = *(const v8bf*)(p);
  u.h[1] = *(const v8bf*)(p + 16);
  return u.v;
}

// ---------------------------------------------------------------- precision prep
__global__ void cvt_f32_bf16(const float* __restrict__ in, __bf16* __restrict__ out, int n) {
  int i = blockIdx.x * blockDim.x + threadIdx.x;
  if (i < n) out[i] = (__bf16)in[i];
}

// in [rows, cols] f32 (row-major)  ->  out [cols, rows] bf16 (row-major) == B^T
__global__ void transpose_to_bf16(const float* __restrict__ in, __bf16* __restrict__ out,
                                  int rows, int cols) {
  int i = blockIdx.x * blockDim.x + threadIdx.x;
  if (i < rows * cols) {
    int r = i / cols, c = i - r * cols;
    out[(size_t)c * rows + r] = (__bf16)in[i];
  }
}

// ---------------------------------------------------------------- QKV projection
// qkv = x @ w_attn + b_attn, scattered to q (scaled 1/8, [B,H,T,D]),
// k ([B,H,T,D]), vT ([B,H,D,T]) all bf16. One wave = 32x64 output tile.
__global__ void gemm_qkv(const __bf16* __restrict__ xb, const __bf16* __restrict__ wT,
                         const float* __restrict__ bias,
                         __bf16* __restrict__ qg, __bf16* __restrict__ kg,
                         __bf16* __restrict__ vTg) {
  const int MT = ROWS / 32;   // 128
  const int NT = C3 / 64;     // 48
  int wave = (blockIdx.x * blockDim.x + threadIdx.x) >> 5;
  int lane = threadIdx.x & 31;
  int mt = wave / NT, nt = wave - mt * NT;
  if (mt >= MT) return;
  int m0 = mt * 32, n0 = nt * 64;

  v8f acc[2][4] = {};
#pragma unroll 4
  for (int k0 = 0; k0 < C_EMB; k0 += 32) {
    v16bf a0 = load_a_frag(xb, C_EMB, m0,      k0, lane);
    v16bf a1 = load_a_frag(xb, C_EMB, m0 + 16, k0, lane);
#pragma unroll
    for (int j = 0; j < 4; ++j) {
      v16bf b = load_bt_frag(wT, C_EMB, n0 + j * 16, k0, lane);
      acc[0][j] = wmma_bf16(a0, b, acc[0][j]);
      acc[1][j] = wmma_bf16(a1, b, acc[1][j]);
    }
  }

  int nlo = lane & 15, half = lane >> 4;
#pragma unroll
  for (int i = 0; i < 2; ++i) {
#pragma unroll
    for (int j = 0; j < 4; ++j) {
      int ncol  = n0 + j * 16 + nlo;
      float bv  = bias[ncol];
      int which = ncol >> 10;       // 0=q, 1=k, 2=v
      int cmod  = ncol & (C_EMB - 1);
      int h = cmod >> 6, d = cmod & (HD - 1);
#pragma unroll
      for (int r = 0; r < 8; ++r) {
        int mrow = m0 + i * 16 + half * 8 + r;
        int bidx = mrow >> 11;        // /SEQ
        int t    = mrow & (SEQ - 1);
        float val = acc[i][j][r] + bv;
        size_t bh = (size_t)(bidx * HEADS + h);
        if (which == 0)
          qg[(bh * SEQ + t) * HD + d] = (__bf16)(val * 0.125f); // 1/sqrt(64)
        else if (which == 1)
          kg[(bh * SEQ + t) * HD + d] = (__bf16)val;
        else
          vTg[(bh * HD + d) * SEQ + t] = (__bf16)val;
      }
    }
  }
}

// ---------------------------------------------------------------- flash attention
// One wave = 16 query rows of one (b,h); loops over 64-key tiles with online softmax.
__global__ void attn_flash(const __bf16* __restrict__ qg, const __bf16* __restrict__ kg,
                           const __bf16* __restrict__ vTg, __bf16* __restrict__ yb) {
  __shared__ __bf16 lds[8][16 * P_LDS_STRIDE];
  int wslot = threadIdx.x >> 5;
  int lane  = threadIdx.x & 31;
  int wave  = blockIdx.x * (blockDim.x >> 5) + wslot;
  const int QT = SEQ / 16;   // 128
  int bh = wave / QT;
  int qt = wave - bh * QT;
  if (bh >= BATCH * HEADS) return;
  int q0 = qt * 16;

  const __bf16* Q  = qg  + (size_t)bh * SEQ * HD;
  const __bf16* K  = kg  + (size_t)bh * SEQ * HD;
  const __bf16* VT = vTg + (size_t)bh * HD * SEQ;

  // Q fragments for both k-steps over D=64 (reused across the whole key loop)
  v16bf qa0 = load_a_frag(Q, HD, q0, 0,  lane);
  v16bf qa1 = load_a_frag(Q, HD, q0, 32, lane);

  v8f y[4] = {};
  v8f mrow, lrow;
#pragma unroll
  for (int r = 0; r < 8; ++r) { mrow[r] = -1e30f; lrow[r] = 0.f; }

  int nlo = lane & 15, half = lane >> 4;
  __bf16* P = &lds[wslot][0];

  for (int j0 = 0; j0 <= q0 + 15; j0 += 64) {
    // ---- S = (Q/8) @ K^T over this 16x64 key tile
    v8f s[4];
#pragma unroll
    for (int j = 0; j < 4; ++j) {
      v16bf b0 = load_bt_frag(K, HD, j0 + j * 16, 0,  lane);
      v16bf b1 = load_bt_frag(K, HD, j0 + j * 16, 32, lane);
      v8f c = {};
      c = wmma_bf16(qa0, b0, c);
      c = wmma_bf16(qa1, b1, c);
      s[j] = c;
    }
    // ---- causal mask only where the tile crosses the diagonal
    if (j0 + 63 > q0) {
#pragma unroll
      for (int j = 0; j < 4; ++j) {
        int kcol = j0 + j * 16 + nlo;
#pragma unroll
        for (int r = 0; r < 8; ++r)
          if (kcol > q0 + half * 8 + r) s[j][r] = -1e30f;
      }
    }
    // ---- online softmax: row max (rows live across 16-lane halves)
    v8f mnew = mrow;
#pragma unroll
    for (int j = 0; j < 4; ++j)
#pragma unroll
      for (int r = 0; r < 8; ++r) mnew[r] = fmaxf(mnew[r], s[j][r]);
#pragma unroll
    for (int mask = 1; mask <= 8; mask <<= 1)
#pragma unroll
      for (int r = 0; r < 8; ++r) mnew[r] = fmaxf(mnew[r], __shfl_xor(mnew[r], mask, 32));

    v8f rsum;
#pragma unroll
    for (int r = 0; r < 8; ++r) rsum[r] = 0.f;
#pragma unroll
    for (int j = 0; j < 4; ++j)
#pragma unroll
      for (int r = 0; r < 8; ++r) {
        float p = __expf(s[j][r] - mnew[r]);
        s[j][r] = p;
        rsum[r] += p;
      }
#pragma unroll
    for (int mask = 1; mask <= 8; mask <<= 1)
#pragma unroll
      for (int r = 0; r < 8; ++r) rsum[r] += __shfl_xor(rsum[r], mask, 32);

#pragma unroll
    for (int r = 0; r < 8; ++r) {
      float sc = __expf(mrow[r] - mnew[r]);
      lrow[r]  = lrow[r] * sc + rsum[r];
      mrow[r]  = mnew[r];
#pragma unroll
      for (int j = 0; j < 4; ++j) y[j][r] *= sc;
    }

    // ---- stage P (C-layout) through LDS, reread as A-fragments (same wave: DS in-order)
#pragma unroll
    for (int j = 0; j < 4; ++j)
#pragma unroll
      for (int r = 0; r < 8; ++r)
        P[(half * 8 + r) * P_LDS_STRIDE + j * 16 + nlo] = (__bf16)s[j][r];

    v16bf pa0 = load_a_frag(P, P_LDS_STRIDE, 0, 0,  lane);
    v16bf pa1 = load_a_frag(P, P_LDS_STRIDE, 0, 32, lane);

    // ---- y += P @ V  (B^T == V^T, contiguous in stored vT)
#pragma unroll
    for (int j = 0; j < 4; ++j) {
      v16bf vb0 = load_bt_frag(VT, SEQ, j * 16, j0,      lane);
      v16bf vb1 = load_bt_frag(VT, SEQ, j * 16, j0 + 32, lane);
      y[j] = wmma_bf16(pa0, vb0, y[j]);
      y[j] = wmma_bf16(pa1, vb1, y[j]);
    }
  }

  // ---- normalize, write y back to [B*T, C] bf16 for the output projection
  int b = bh >> 4, h = bh & (HEADS - 1);
#pragma unroll
  for (int j = 0; j < 4; ++j)
#pragma unroll
    for (int r = 0; r < 8; ++r) {
      int t = q0 + half * 8 + r;
      float val = y[j][r] / lrow[r];
      yb[((size_t)b * SEQ + t) * C_EMB + h * HD + j * 16 + nlo] = (__bf16)val;
    }
}

// ---------------------------------------------------------------- output projection
__global__ void gemm_proj(const __bf16* __restrict__ yb, const __bf16* __restrict__ wT,
                          const float* __restrict__ bias, float* __restrict__ out) {
  const int MT = ROWS / 32;    // 128
  const int NT = C_EMB / 64;   // 16
  int wave = (blockIdx.x * blockDim.x + threadIdx.x) >> 5;
  int lane = threadIdx.x & 31;
  int mt = wave / NT, nt = wave - mt * NT;
  if (mt >= MT) return;
  int m0 = mt * 32, n0 = nt * 64;

  v8f acc[2][4] = {};
#pragma unroll 4
  for (int k0 = 0; k0 < C_EMB; k0 += 32) {
    v16bf a0 = load_a_frag(yb, C_EMB, m0,      k0, lane);
    v16bf a1 = load_a_frag(yb, C_EMB, m0 + 16, k0, lane);
#pragma unroll
    for (int j = 0; j < 4; ++j) {
      v16bf b = load_bt_frag(wT, C_EMB, n0 + j * 16, k0, lane);
      acc[0][j] = wmma_bf16(a0, b, acc[0][j]);
      acc[1][j] = wmma_bf16(a1, b, acc[1][j]);
    }
  }

  int nlo = lane & 15, half = lane >> 4;
#pragma unroll
  for (int i = 0; i < 2; ++i)
#pragma unroll
    for (int j = 0; j < 4; ++j) {
      int ncol = n0 + j * 16 + nlo;
      float bv = bias[ncol];
#pragma unroll
      for (int r = 0; r < 8; ++r) {
        int mrow = m0 + i * 16 + half * 8 + r;
        out[(size_t)mrow * C_EMB + ncol] = acc[i][j][r] + bv;
      }
    }
}

// ---------------------------------------------------------------- launch
extern "C" void kernel_launch(void* const* d_in, const int* in_sizes, int n_in,
                              void* d_out, int out_size, void* d_ws, size_t ws_size,
                              hipStream_t stream) {
  const float* x      = (const float*)d_in[0];
  const float* w_attn = (const float*)d_in[1];
  const float* b_attn = (const float*)d_in[2];
  const float* w_proj = (const float*)d_in[3];
  const float* b_proj = (const float*)d_in[4];
  float* out = (float*)d_out;

  char* ws = (char*)d_ws;
  size_t off = 0;
  __bf16* xb   = (__bf16*)(ws + off); off += (size_t)ROWS * C_EMB * 2;     // 8 MB
  __bf16* waT  = (__bf16*)(ws + off); off += (size_t)C3 * C_EMB * 2;       // 6 MB
  __bf16* wpT  = (__bf16*)(ws + off); off += (size_t)C_EMB * C_EMB * 2;    // 2 MB
  __bf16* qg   = (__bf16*)(ws + off); off += (size_t)ROWS * C_EMB * 2;     // 8 MB
  __bf16* kg   = (__bf16*)(ws + off); off += (size_t)ROWS * C_EMB * 2;     // 8 MB
  __bf16* vTg  = (__bf16*)(ws + off); off += (size_t)ROWS * C_EMB * 2;     // 8 MB
  __bf16* yb   = (__bf16*)(ws + off); off += (size_t)ROWS * C_EMB * 2;     // 8 MB
  (void)ws_size; (void)in_sizes; (void)n_in; (void)out_size;

  int nx = ROWS * C_EMB;
  cvt_f32_bf16<<<(nx + 255) / 256, 256, 0, stream>>>(x, xb, nx);
  int nwa = C_EMB * C3;
  transpose_to_bf16<<<(nwa + 255) / 256, 256, 0, stream>>>(w_attn, waT, C_EMB, C3);
  int nwp = C_EMB * C_EMB;
  transpose_to_bf16<<<(nwp + 255) / 256, 256, 0, stream>>>(w_proj, wpT, C_EMB, C_EMB);

  {
    int waves = (ROWS / 32) * (C3 / 64);            // 6144
    gemm_qkv<<<waves / 8, 256, 0, stream>>>(xb, waT, b_attn, qg, kg, vTg);
  }
  {
    int waves = BATCH * HEADS * (SEQ / 16);         // 4096
    attn_flash<<<waves / 8, 256, 0, stream>>>(qg, kg, vTg, yb);
  }
  {
    int waves = (ROWS / 32) * (C_EMB / 64);         // 2048
    gemm_proj<<<waves / 8, 256, 0, stream>>>(yb, wpT, b_proj, out);
  }
}
